// ReverseLSTMLayer_36532991820402
// MI455X (gfx1250) — matrix operand order repaired and checked by reading, not compiled, not run on hardware
//
#include <hip/hip_runtime.h>
#include <hip/hip_bf16.h>
#include <stdint.h>

typedef __bf16 bf16_t;
typedef __attribute__((ext_vector_type(16))) __bf16 v16bf;
typedef __attribute__((ext_vector_type(8)))  __bf16 v8bf;
typedef __attribute__((ext_vector_type(8)))  float  v8f;

#define T_DIM 512
#define B_DIM 64
#define I_DIM 1024
#define H_DIM 1024
#define G4_DIM 4096   // 4*H
#define NBLOCKS 32    // persistent grid

__device__ __forceinline__ float sigmoidf_(float x) { return 1.0f / (1.0f + __expf(-x)); }

// A fragment (16x32 bf16): lane%16 = M row; +16 K for upper half-wave;
// VGPR 0-7 = K kb..kb+15 (contiguous, matches B)  -> two 16B loads.
__device__ __forceinline__ v16bf load_frag_a(const bf16_t* __restrict__ src, int ld,
                                             int m0, int k0, int lane) {
    int row = m0 + (lane & 15);
    int kb  = k0 + ((lane >> 4) << 4);
    const bf16_t* p = src + (size_t)row * ld + kb;
    v8bf lo = *(const v8bf*)(p);
    v8bf hi = *(const v8bf*)(p + 8);
    return __builtin_shufflevector(lo, hi, 0,1,2,3,4,5,6,7,8,9,10,11,12,13,14,15);
}

// B fragment (32x16) with B = W^T, W row-major: B[k][n] = W[n0+n][k0+k].
// lane%16 = N col; K 0-15 (lanes 0-15) / 16-31 (lanes 16-31): 16 contiguous bf16 -> one 32B load.
__device__ __forceinline__ v16bf load_frag_b(const bf16_t* __restrict__ w, int ldw,
                                             int n0, int k0, int lane) {
    int row = n0 + (lane & 15);
    int kb  = k0 + ((lane >> 4) << 4);
    return *(const v16bf*)(w + (size_t)row * ldw + kb);
}

// ---------------- prep: bf16 conversions, bias fold, state + barrier init ----------------
__global__ void prep_kernel(const float* __restrict__ w_ih, const float* __restrict__ w_hh,
                            const float* __restrict__ b_ih, const float* __restrict__ b_hh,
                            bf16_t* __restrict__ wih_b, bf16_t* __restrict__ whh_b,
                            float* __restrict__ bias,
                            bf16_t* __restrict__ h0, bf16_t* __restrict__ h1,
                            unsigned int* __restrict__ bar) {
    int i = blockIdx.x * blockDim.x + threadIdx.x;
    if (i < G4_DIM * I_DIM) {
        wih_b[i] = (bf16_t)w_ih[i];
        whh_b[i] = (bf16_t)w_hh[i];
    }
    if (i < G4_DIM) bias[i] = b_ih[i] + b_hh[i];
    if (i < B_DIM * H_DIM) {
        h0[i] = (bf16_t)0.0f;
        h1[i] = (bf16_t)0.0f;
    }
    if (i == 0) *bar = 0u;   // reset device barrier every call (ws is not re-poisoned)
}

__global__ void cvt_x_kernel(const float* __restrict__ x, bf16_t* __restrict__ xb, int n) {
    int i = blockIdx.x * blockDim.x + threadIdx.x;
    if (i < n) xb[i] = (bf16_t)x[i];
}

// ---------------- input projection GEMM: G = Xb @ Wih^T + bias  (32768 x 4096 x 1024) ----
__global__ void __launch_bounds__(256) input_gemm_kernel(const bf16_t* __restrict__ xb,
                                                         const bf16_t* __restrict__ wih,
                                                         const float*  __restrict__ bias,
                                                         float*        __restrict__ G) {
    const int wave = threadIdx.x >> 5;
    const int lane = threadIdx.x & 31;
    const int idx  = blockIdx.x * 8 + wave;
    const int nStrips = G4_DIM / 64;                // 64
    const int m0 = (idx / nStrips) * 16;
    const int n0 = (idx % nStrips) * 64;

    v8f acc[4] = {};
    for (int k0 = 0; k0 < I_DIM; k0 += 32) {
        v16bf a = load_frag_a(xb, I_DIM, m0, k0, lane);
#pragma unroll
        for (int q = 0; q < 4; ++q) {
            v16bf b = load_frag_b(wih, I_DIM, n0 + q * 16, k0, lane);
            acc[q] = __builtin_amdgcn_wmma_f32_16x16x32_bf16(false, a, false, b,
                                                             (short)0, acc[q], false, false);
        }
    }
    const int hi8 = (lane >> 4) << 3;
#pragma unroll
    for (int q = 0; q < 4; ++q) {
        int col = n0 + q * 16 + (lane & 15);
        float bcol = bias[col];
#pragma unroll
        for (int r = 0; r < 8; ++r) {
            int row = m0 + r + hi8;
            G[(size_t)row * G4_DIM + col] = acc[q][r] + bcol;
        }
    }
}

// ---------------- persistent recurrent kernel: all 512 steps in one launch --------------
// 32 blocks x 8 waves. Block bx: batch tile m0 = (bx&3)*16, column group nb0 = (bx>>2)*128
// (8 waves cover nb0..nb0+127 within each of the 4 gate sections).
// c state lives in VGPRs for the whole scan; h double-buffered in global, staged via LDS;
// G[t] tiles pulled into LDS with async-to-LDS loads overlapping the WMMA K-loop.
__global__ void __launch_bounds__(256) lstm_persistent_kernel(
        const bf16_t* __restrict__ whh,
        const float*  __restrict__ G,
        float*        __restrict__ ys,          // (T,B,H)
        bf16_t*       __restrict__ h0,
        bf16_t*       __restrict__ h1,
        float*        __restrict__ out_h,
        float*        __restrict__ out_c,
        unsigned int* __restrict__ bar) {
    __shared__ bf16_t lds_h[16 * H_DIM];        // 32 KB: h rows m0..m0+15
    __shared__ float  lds_g[4 * 16 * 128];      // 32 KB: G[t] tiles for this block

    const int tid  = threadIdx.x;
    const int wave = tid >> 5;
    const int lane = tid & 31;
    const int bx   = blockIdx.x;
    const int m0   = (bx & 3) * 16;
    const int nb0  = (bx >> 2) * 128;
    const int n0   = nb0 + wave * 16;
    const int hi8  = (lane >> 4) << 3;
    const int col  = n0 + (lane & 15);          // column within gate section / h

    // per-thread slice of the h copy: 64 bf16 (128B) contiguous
    const int hrow = tid >> 4;                  // 0..15
    const int hcol = (tid & 15) * 64;
    // per-thread slice of the Gt async copy: 32 floats (128B)
    const int grow  = tid & 15;                 // 0..15
    const int gseg  = tid >> 4;                 // 0..15
    const int ggate = gseg >> 2;                // 0..3
    const int gchk  = (gseg & 3) * 32;          // 0,32,64,96

    // LDS byte offsets (generic shared address low 32 bits = LDS offset)
    const unsigned int lds_g_dst = (unsigned int)(uintptr_t)&lds_g[(ggate * 16 + grow) * 128 + gchk];

    v8f creg = {};                              // cell state, registers across all steps

    for (int s = 0; s < T_DIM; ++s) {
        const int t = T_DIM - 1 - s;
        const bf16_t* hin  = (s & 1) ? h1 : h0;
        bf16_t*       hout = (s & 1) ? h0 : h1;
        const float*  Gt   = G + (size_t)t * B_DIM * G4_DIM;

        // kick async copy of this block's G[t] tiles into LDS; overlaps the whole GEMM
        {
            const float* gsrc = Gt + (size_t)(m0 + grow) * G4_DIM + ggate * H_DIM + nb0 + gchk;
#pragma unroll
            for (int j = 0; j < 8; ++j) {
                uint64_t     ga = (uint64_t)(uintptr_t)(gsrc + j * 4);
                unsigned int la = lds_g_dst + (unsigned)(j * 16);
                asm volatile("global_load_async_to_lds_b128 %0, %1, off"
                             :: "v"(la), "v"(ga) : "memory");
            }
        }

        // cooperative copy of h rows m0..m0+15 into LDS (global_load -> ds_store)
        {
            const bf16_t* hsrc = hin + (size_t)(m0 + hrow) * H_DIM + hcol;
            bf16_t*       hdst = &lds_h[hrow * H_DIM + hcol];
#pragma unroll
            for (int j = 0; j < 8; ++j)
                *(v8bf*)(hdst + j * 8) = *(const v8bf*)(hsrc + j * 8);
        }
        __syncthreads();                        // lds_h ready

        v8f acc[4] = {};
        for (int k0 = 0; k0 < H_DIM; k0 += 32) {
            v16bf a = load_frag_a(lds_h, H_DIM, 0, k0, lane);   // ds_load_b128 x2
#pragma unroll
            for (int q = 0; q < 4; ++q) {
                v16bf b = load_frag_b(whh, H_DIM, q * H_DIM + n0, k0, lane);
                acc[q] = __builtin_amdgcn_wmma_f32_16x16x32_bf16(false, a, false, b,
                                                                 (short)0, acc[q], false, false);
            }
        }

        asm volatile("s_wait_asynccnt 0x0" ::: "memory");
        __syncthreads();                        // lds_g ready for all waves

        // fused gate activations + cell update epilogue (c in registers)
        const int gcolL = wave * 16 + (lane & 15);   // col within block's 128-wide group
#pragma unroll
        for (int r = 0; r < 8; ++r) {
            int brow = r + hi8;                      // 0..15
            int b    = m0 + brow;
            float iv = sigmoidf_(acc[0][r] + lds_g[(0 * 16 + brow) * 128 + gcolL]);
            float fv = sigmoidf_(acc[1][r] + lds_g[(1 * 16 + brow) * 128 + gcolL]);
            float gv = tanhf    (acc[2][r] + lds_g[(2 * 16 + brow) * 128 + gcolL]);
            float ov = sigmoidf_(acc[3][r] + lds_g[(3 * 16 + brow) * 128 + gcolL]);
            float cv = fv * creg[r] + iv * gv;
            float hv = ov * tanhf(cv);
            creg[r] = cv;
            size_t oi = (size_t)b * H_DIM + col;
            ys[(size_t)t * B_DIM * H_DIM + oi] = hv;
            hout[oi] = (bf16_t)hv;
            if (t == 0) out_h[oi] = hv;              // final carry h == ys[0]
        }

        // device-wide barrier: release hout, all-arrive, acquire
        __threadfence();
        __syncthreads();
        if (tid == 0) {
            __hip_atomic_fetch_add(bar, 1u, __ATOMIC_RELEASE, __HIP_MEMORY_SCOPE_AGENT);
            unsigned int target = (unsigned)NBLOCKS * (unsigned)(s + 1);
            while (__hip_atomic_load(bar, __ATOMIC_ACQUIRE, __HIP_MEMORY_SCOPE_AGENT) < target)
                __builtin_amdgcn_s_sleep(1);
        }
        __syncthreads();
    }

    // spill final cell state
#pragma unroll
    for (int r = 0; r < 8; ++r) {
        int b = m0 + r + hi8;
        out_c[(size_t)b * H_DIM + col] = creg[r];
    }
}

extern "C" void kernel_launch(void* const* d_in, const int* in_sizes, int n_in,
                              void* d_out, int out_size, void* d_ws, size_t ws_size,
                              hipStream_t stream) {
    const float* x    = (const float*)d_in[0];
    const float* w_ih = (const float*)d_in[1];
    const float* w_hh = (const float*)d_in[2];
    const float* b_ih = (const float*)d_in[3];
    const float* b_hh = (const float*)d_in[4];
    float* out = (float*)d_out;

    uint8_t* ws = (uint8_t*)d_ws;
    auto alignup = [](size_t o) { return (o + 255) & ~(size_t)255; };
    size_t off = 0;
    bf16_t* wih_b = (bf16_t*)(ws + off); off = alignup(off + (size_t)G4_DIM * I_DIM * sizeof(bf16_t));
    bf16_t* whh_b = (bf16_t*)(ws + off); off = alignup(off + (size_t)G4_DIM * H_DIM * sizeof(bf16_t));
    bf16_t* xb    = (bf16_t*)(ws + off); off = alignup(off + (size_t)T_DIM * B_DIM * I_DIM * sizeof(bf16_t));
    float*  bias  = (float*) (ws + off); off = alignup(off + (size_t)G4_DIM * sizeof(float));
    bf16_t* hbuf0 = (bf16_t*)(ws + off); off = alignup(off + (size_t)B_DIM * H_DIM * sizeof(bf16_t));
    bf16_t* hbuf1 = (bf16_t*)(ws + off); off = alignup(off + (size_t)B_DIM * H_DIM * sizeof(bf16_t));
    unsigned int* bar = (unsigned int*)(ws + off); off = alignup(off + 256);
    float*  G     = (float*) (ws + off); off = alignup(off + (size_t)T_DIM * B_DIM * G4_DIM * sizeof(float));

    // 1) conversions + bias fold + state/barrier init
    {
        int n = G4_DIM * I_DIM;
        prep_kernel<<<(n + 255) / 256, 256, 0, stream>>>(w_ih, w_hh, b_ih, b_hh,
                                                         wih_b, whh_b, bias,
                                                         hbuf0, hbuf1, bar);
    }
    {
        int n = T_DIM * B_DIM * I_DIM;
        cvt_x_kernel<<<(n + 255) / 256, 256, 0, stream>>>(x, xb, n);
    }

    // 2) parallel input projection: G = Xb @ Wih^T + bias
    {
        int strips = (T_DIM * B_DIM / 16) * (G4_DIM / 64);  // 131072 waves
        input_gemm_kernel<<<strips / 8, 256, 0, stream>>>(xb, wih_b, bias, G);
    }

    // 3) single persistent kernel runs all 512 dependent steps
    {
        float* out_h = out + (size_t)T_DIM * B_DIM * H_DIM;
        float* out_c = out_h + (size_t)B_DIM * H_DIM;
        lstm_persistent_kernel<<<NBLOCKS, 256, 0, stream>>>(whh_b, G, out,
                                                            hbuf0, hbuf1,
                                                            out_h, out_c, bar);
    }
}
